// WindowAttention_56143812494069
// MI455X (gfx1250) — matrix operand order repaired and checked by reading, not compiled, not run on hardware
//
#include <hip/hip_runtime.h>

typedef __attribute__((ext_vector_type(16))) _Float16 v16h;
typedef __attribute__((ext_vector_type(8)))  float    v8f;

#define NTOK 49
#define CDIM 192
#define NHEAD 6
#define HD 32
#define MPAD 64

union Frag { v16h h; unsigned int u[8]; uint4 q4[2]; };

// A-operand 16x32 f16: row = lane&15; lane<16 -> K 0..15, lane>=16 -> K 16..31
// Two 16B loads (offsets 0 and +8 halves), both 16B-aligned for ldh multiple of 8.
__device__ __forceinline__ v16h frag_a(const _Float16* __restrict__ p, int ldh) {
  const int l = threadIdx.x & 31;
  const _Float16* q = p + (l & 15) * ldh + ((l >> 4) << 4);
  Frag f;
  f.q4[0] = *(const uint4*)(q);
  f.q4[1] = *(const uint4*)(q + 8);
  return f.h;
}

// B-operand 32x16 f16: col = lane&15; lane<16 -> K 0..15, lane>=16 -> K 16..31.
// 8 consecutive u32 per lane -> two 16B loads.
__device__ __forceinline__ v16h frag_b(const _Float16* __restrict__ p, int ldh) {
  const int l = threadIdx.x & 31;
  const _Float16* q = p + (l & 15) * ldh + ((l >> 4) << 4);
  Frag f;
  f.q4[0] = *(const uint4*)(q);
  f.q4[1] = *(const uint4*)(q + 8);
  return f.h;
}

__device__ __forceinline__ unsigned int pk2(float a, float b) {
  unsigned short lo = __builtin_bit_cast(unsigned short, (_Float16)a);
  unsigned short hi = __builtin_bit_cast(unsigned short, (_Float16)b);
  return (unsigned int)lo | ((unsigned int)hi << 16);
}

// Fallback B-operand from f32 row-major weights (convert in-register)
__device__ __forceinline__ v16h frag_b_f32(const float* __restrict__ p, int ldk) {
  const int l = threadIdx.x & 31;
  const float* q = p + (l & 15) * ldk + ((l >> 4) << 4);
  Frag f;
#pragma unroll
  for (int i = 0; i < 4; ++i) {
    float4 v = *(const float4*)(q + 4 * i);
    f.u[2 * i]     = pk2(v.x, v.y);
    f.u[2 * i + 1] = pk2(v.z, v.w);
  }
  return f.h;
}

__device__ __forceinline__ v8f wmma16(v16h a, v16h b, v8f c) {
  return __builtin_amdgcn_wmma_f32_16x16x32_f16(false, a, false, b, (short)0, c, false, false);
}

// One-shot f32->f16 weight conversion into workspace (shared by all 4096 blocks)
__global__ __launch_bounds__(256)
void convert_weights(const float* __restrict__ qkv_w, const float* __restrict__ proj_w,
                     _Float16* __restrict__ w16) {
  const int i = blockIdx.x * 256 + threadIdx.x;
  const int nq = 3 * CDIM * CDIM;            // 110592
  const int nt = nq + CDIM * CDIM;           // 147456
  if (i < nq)      w16[i] = (_Float16)qkv_w[i];
  else if (i < nt) w16[i] = (_Float16)proj_w[i - nq];
}

template <bool W16>
__global__ __launch_bounds__(256)
void swin_window_attn(const float* __restrict__ x, const float* __restrict__ mask,
                      const float* __restrict__ rpb, const int* __restrict__ rix,
                      const float* __restrict__ qkv_w, const float* __restrict__ qkv_b,
                      const float* __restrict__ proj_w, const float* __restrict__ proj_b,
                      const _Float16* __restrict__ qkv_w16,
                      const _Float16* __restrict__ proj_w16,
                      float* __restrict__ out) {
  __shared__ __align__(16) _Float16 s_q [MPAD * CDIM];   // Q (scaled) row-major
  __shared__ __align__(16) _Float16 s_k [MPAD * CDIM];   // K row-major (contiguous d -> B-operand)
  __shared__ __align__(16) _Float16 s_vT[CDIM * MPAD];   // V transposed (contiguous tokens -> B-operand)
  __shared__ __align__(16) _Float16 s_xo[MPAD * CDIM];   // x (phase1 A), later attn-out (phase4 A)
  __shared__ __align__(16) float    s_s [MPAD * MPAD];   // f32 scores, one head at a time
  __shared__ __align__(16) _Float16 s_p [MPAD * MPAD];   // f16 probabilities

  const int b     = blockIdx.x;
  const int tid   = threadIdx.x;
  const int wave  = tid >> 5;
  const int lane  = tid & 31;
  const int col16 = lane & 15;
  const int hi8   = (lane >> 4) << 3;
  const float scale = 0.17677669529663687f;  // 1/sqrt(32)

  // Warm WGP$/L2 with the (shared) weights
  if constexpr (W16) {
    for (int i = tid; i < 4 * CDIM; i += 256) __builtin_prefetch(qkv_w16 + i * CDIM, 0, 1);
  } else {
    for (int i = tid; i < 3 * CDIM; i += 256) __builtin_prefetch(qkv_w + i * CDIM, 0, 1);
    for (int i = tid; i < CDIM; i += 256)     __builtin_prefetch(proj_w + i * CDIM, 0, 1);
  }

  // ---- Phase 0: x -> LDS f16 (float4 in, packed uint2 out), zero-pad rows 49..63 ----
  const float* xb = x + (size_t)b * (NTOK * CDIM);
  uint2* xo64 = (uint2*)s_xo;
  for (int i = tid; i < MPAD * CDIM / 4; i += 256) {
    int e = 4 * i;
    uint2 v = make_uint2(0u, 0u);
    if (e < NTOK * CDIM) {
      float4 f = *(const float4*)(xb + e);
      v.x = pk2(f.x, f.y);
      v.y = pk2(f.z, f.w);
    }
    xo64[i] = v;
  }
  __syncthreads();

  // ---- Phase 1: QKV = x @ W^T + b  (144 tiles, 6 K-steps each) ----
  for (int t = wave; t < 144; t += 8) {
    const int mt = t & 3;       // token tile
    const int nt = t >> 2;      // 0..11 Q, 12..23 K, 24..35 V
    const float bv = qkv_b[nt * 16 + col16];
    v8f acc = {bv, bv, bv, bv, bv, bv, bv, bv};
#pragma unroll
    for (int kk = 0; kk < 6; ++kk) {
      v16h a = frag_a(s_xo + mt * 16 * CDIM + kk * 32, CDIM);
      v16h bf;
      if constexpr (W16) bf = frag_b(qkv_w16 + nt * 16 * CDIM + kk * 32, CDIM);
      else               bf = frag_b_f32(qkv_w + nt * 16 * CDIM + kk * 32, CDIM);
      acc = wmma16(a, bf, acc);
    }
    if (nt < 12) {
#pragma unroll
      for (int r = 0; r < 8; ++r)
        s_q[(mt * 16 + hi8 + r) * CDIM + nt * 16 + col16] = (_Float16)(acc[r] * scale);
    } else if (nt < 24) {
#pragma unroll
      for (int r = 0; r < 8; ++r)
        s_k[(mt * 16 + hi8 + r) * CDIM + (nt - 12) * 16 + col16] = (_Float16)acc[r];
    } else {
      // V transposed: 8 consecutive rows for fixed column -> one 16B DS store
      union { uint4 v4; unsigned int u[4]; } pkd;
#pragma unroll
      for (int r = 0; r < 4; ++r) pkd.u[r] = pk2(acc[2 * r], acc[2 * r + 1]);
      *(uint4*)(&s_vT[((nt - 24) * 16 + col16) * MPAD + mt * 16 + hi8]) = pkd.v4;
    }
  }
  __syncthreads();

  const float* maskb = mask + (size_t)(b & 63) * (NTOK * NTOK);
  float* ob = out + (size_t)b * (NTOK * CDIM);

  for (int h = 0; h < NHEAD; ++h) {
    // ---- Phase 2: S = Qs @ K^T + rel_bias + mask (bias folded into accumulator init) ----
#pragma unroll
    for (int tt = 0; tt < 2; ++tt) {
      const int t  = wave * 2 + tt;
      const int mt = t >> 2;
      const int nt = t & 3;
      const int j  = nt * 16 + col16;
      v8f acc;
#pragma unroll
      for (int r = 0; r < 8; ++r) {
        const int m = mt * 16 + hi8 + r;
        float c0 = 0.0f;
        if (m < NTOK && j < NTOK)
          c0 = rpb[rix[m * NTOK + j] * NHEAD + h] + maskb[m * NTOK + j];
        acc[r] = c0;
      }
      v16h a  = frag_a(s_q + mt * 16 * CDIM + h * HD, CDIM);
      v16h bf = frag_b(s_k + nt * 16 * CDIM + h * HD, CDIM);
      acc = wmma16(a, bf, acc);   // K=32 == head dim: one WMMA per score tile
#pragma unroll
      for (int r = 0; r < 8; ++r)
        s_s[(mt * 16 + hi8 + r) * MPAD + nt * 16 + col16] = acc[r];
    }
    __syncthreads();

    // ---- Softmax: 4 threads per row, shfl_xor(width=4) reductions ----
    {
      const int srow = tid >> 2;
      const int ssub = tid & 3;
      if (srow < NTOK) {
        float mx = -3.0e38f;
        for (int j = ssub; j < NTOK; j += 4) mx = fmaxf(mx, s_s[srow * MPAD + j]);
        mx = fmaxf(mx, __shfl_xor(mx, 1, 4));
        mx = fmaxf(mx, __shfl_xor(mx, 2, 4));
        float sum = 0.0f;
        for (int j = ssub; j < NTOK; j += 4) {
          float e = __expf(s_s[srow * MPAD + j] - mx);
          s_s[srow * MPAD + j] = e; sum += e;
        }
        sum += __shfl_xor(sum, 1, 4);
        sum += __shfl_xor(sum, 2, 4);
        const float inv = 1.0f / sum;
        for (int j = ssub; j < NTOK; j += 4)
          s_p[srow * MPAD + j] = (_Float16)(s_s[srow * MPAD + j] * inv);
        for (int j = NTOK + ssub; j < MPAD; j += 4)
          s_p[srow * MPAD + j] = (_Float16)0.0f;
      } else {
        for (int j = ssub; j < MPAD; j += 4)
          s_p[srow * MPAD + j] = (_Float16)0.0f;
      }
    }
    __syncthreads();

    // ---- Phase 3: O_h = P @ V_h into s_xo cols [h*32, h*32+32) ----
    {
      const int mt = wave >> 1;
      const int nt = wave & 1;
      v8f acc = {0, 0, 0, 0, 0, 0, 0, 0};
#pragma unroll
      for (int kk = 0; kk < 2; ++kk) {
        v16h a  = frag_a(s_p + mt * 16 * MPAD + kk * 32, MPAD);
        v16h bf = frag_b(s_vT + (h * HD + nt * 16) * MPAD + kk * 32, MPAD);
        acc = wmma16(a, bf, acc);
      }
#pragma unroll
      for (int r = 0; r < 8; ++r)
        s_xo[(mt * 16 + hi8 + r) * CDIM + h * HD + nt * 16 + col16] = (_Float16)acc[r];
    }
    // s_s is rewritten only after the next barrier; s_p only after two barriers -> race-free
  }
  __syncthreads();

  // ---- Phase 4: out = O @ proj_w^T + proj_b ----
  for (int t = wave; t < 48; t += 8) {
    const int mt = t & 3;
    const int nt = t >> 2;
    const float bv = proj_b[nt * 16 + col16];
    v8f acc = {bv, bv, bv, bv, bv, bv, bv, bv};
#pragma unroll
    for (int kk = 0; kk < 6; ++kk) {
      v16h a = frag_a(s_xo + mt * 16 * CDIM + kk * 32, CDIM);
      v16h bf;
      if constexpr (W16) bf = frag_b(proj_w16 + nt * 16 * CDIM + kk * 32, CDIM);
      else               bf = frag_b_f32(proj_w + nt * 16 * CDIM + kk * 32, CDIM);
      acc = wmma16(a, bf, acc);
    }
#pragma unroll
    for (int r = 0; r < 8; ++r) {
      int m = mt * 16 + hi8 + r;
      if (m < NTOK) ob[m * CDIM + nt * 16 + col16] = acc[r];
    }
  }
}

extern "C" void kernel_launch(void* const* d_in, const int* in_sizes, int n_in,
                              void* d_out, int out_size, void* d_ws, size_t ws_size,
                              hipStream_t stream) {
  (void)in_sizes; (void)n_in; (void)out_size;
  const float* x      = (const float*)d_in[0];
  const float* mask   = (const float*)d_in[1];
  const float* rpb    = (const float*)d_in[2];
  const int*   rix    = (const int*)d_in[3];
  const float* qkv_w  = (const float*)d_in[4];
  const float* qkv_b  = (const float*)d_in[5];
  const float* proj_w = (const float*)d_in[6];
  const float* proj_b = (const float*)d_in[7];
  float* out = (float*)d_out;

  const size_t w16_bytes = (size_t)(4 * CDIM * CDIM) * sizeof(_Float16);  // 294912
  dim3 grid(4096), block(256);

  if (d_ws != nullptr && ws_size >= w16_bytes) {
    _Float16* w16 = (_Float16*)d_ws;
    convert_weights<<<dim3(4 * CDIM * CDIM / 256), block, 0, stream>>>(qkv_w, proj_w, w16);
    swin_window_attn<true><<<grid, block, 0, stream>>>(
        x, mask, rpb, rix, qkv_w, qkv_b, proj_w, proj_b,
        w16, w16 + 3 * CDIM * CDIM, out);
  } else {
    swin_window_attn<false><<<grid, block, 0, stream>>>(
        x, mask, rpb, rix, qkv_w, qkv_b, proj_w, proj_b,
        nullptr, nullptr, out);
  }
}
